// Attention_1_71519795413096
// MI455X (gfx1250) — matrix operand order repaired and checked by reading, not compiled, not run on hardware
//
#include <hip/hip_runtime.h>

// ---------------------------------------------------------------------------
// Complex attention for MI455X (gfx1250, wave32, WMMA f32_16x16x32_f16).
// B=16, S=512, E=32, D=E*E=1024.
// Pipeline: proj (WMMA) -> scores (WMMA) -> softmax -> context (WMMA).
// Workspace use: 6*16MB f16 q/k/v + 16MB f32 scores + 8MB f16 probs = 120MB.
// ---------------------------------------------------------------------------

typedef __attribute__((ext_vector_type(16))) _Float16 v16h;
typedef __attribute__((ext_vector_type(8)))  float    v8f;

#define NB 16
#define NS 512
#define NE 32
#define ND 1024

union Frag16 {
    v16h   v;
    float4 q[2];   // 2 x 16B = 32B = 16 halves
};

// D = A*B + C, 16x16x32 f16 -> f32
__device__ __forceinline__ v8f wmma_f16(v16h a, v16h b, v8f c) {
    return __builtin_amdgcn_wmma_f32_16x16x32_f16(
        /*neg_a=*/false, a, /*neg_b=*/false, b,
        /*c_mod=*/(short)0, c, /*reuse_a=*/false, /*reuse_b=*/false);
}

// A-matrix fragment (16x32 f16), row-major source with row stride ld (halves).
// ISA layout: lane[0..15] M=lane, K={0..7,16..23}; lane[16..31] M=lane-16,
// K={8..15,24..31}. Two 16B vector loads per lane.
__device__ __forceinline__ v16h load_a_f16(const _Float16* base, int ld, int lane) {
    const int m  = lane & 15;
    const int kb = (lane & 16) ? 8 : 0;
    const _Float16* p = base + (size_t)m * ld + kb;
    Frag16 f;
    f.q[0] = *reinterpret_cast<const float4*>(p);        // K = kb + 0..7
    f.q[1] = *reinterpret_cast<const float4*>(p + 16);   // K = kb + 16..23
    return f.v;
}

// B-matrix fragment (32x16 f16): element (K,N) = base[N*ld + K].
// Lane holds col N=lane&15; K={0..7,16..23} (lanes 0-15) / {8..15,24..31} (16-31):
// two 16B loads per lane.
__device__ __forceinline__ v16h load_b_f16(const _Float16* base, int ld, int lane) {
    const int n  = lane & 15;
    const int kb = (lane & 16) ? 8 : 0;
    const _Float16* p = base + (size_t)n * ld + kb;
    Frag16 f;
    f.q[0] = *reinterpret_cast<const float4*>(p);
    f.q[1] = *reinterpret_cast<const float4*>(p + 16);
    return f.v;
}

// A fragment from f32 source (convert to f16 in registers).
__device__ __forceinline__ v16h load_a_f32(const float* base, int ld, int lane) {
    const int m  = lane & 15;
    const int kb = (lane & 16) ? 8 : 0;
    const float* p = base + (size_t)m * ld + kb;
    float4 c0 = *reinterpret_cast<const float4*>(p);
    float4 c1 = *reinterpret_cast<const float4*>(p + 4);
    float4 c2 = *reinterpret_cast<const float4*>(p + 16);
    float4 c3 = *reinterpret_cast<const float4*>(p + 20);
    v16h a;
    a[0] = (_Float16)c0.x; a[1] = (_Float16)c0.y; a[2] = (_Float16)c0.z; a[3] = (_Float16)c0.w;
    a[4] = (_Float16)c1.x; a[5] = (_Float16)c1.y; a[6] = (_Float16)c1.z; a[7] = (_Float16)c1.w;
    a[8]  = (_Float16)c2.x; a[9]  = (_Float16)c2.y; a[10] = (_Float16)c2.z; a[11] = (_Float16)c2.w;
    a[12] = (_Float16)c3.x; a[13] = (_Float16)c3.y; a[14] = (_Float16)c3.z; a[15] = (_Float16)c3.w;
    return a;
}

// B fragment from 32x32 f32 weight (col tile n0): element (K,N)=W[K*32+n0+N].
__device__ __forceinline__ v16h load_b_w32(const float* W, int n0, int lane) {
    const int n  = n0 + (lane & 15);
    const int kb = (lane & 16) ? 8 : 0;
    v16h b;
#pragma unroll
    for (int i = 0; i < 16; ++i) b[i] = (_Float16)W[(kb + (i < 8 ? i : i + 8)) * 32 + n];
    return b;
}

// ---------------------------------------------------------------------------
// Kernel 1: complex projections.  M = B*S*E = 262144 rows of 32.
// grid.y selects projection: 0=q (Q_r,Q_i,Wq), 1=k (KV,Wk), 2=v (KV,Wv).
// q,k stored row-major f16 [B,S,D]; v stored transposed f16 [B,D,S].
// ---------------------------------------------------------------------------
__global__ void proj_kernel(const float* __restrict__ Qr, const float* __restrict__ Qi,
                            const float* __restrict__ KVr, const float* __restrict__ KVi,
                            const float* __restrict__ Wqr, const float* __restrict__ Wqi,
                            const float* __restrict__ Wkr, const float* __restrict__ Wki,
                            const float* __restrict__ Wvr, const float* __restrict__ Wvi,
                            _Float16* __restrict__ qfr, _Float16* __restrict__ qfi,
                            _Float16* __restrict__ kfr, _Float16* __restrict__ kfi,
                            _Float16* __restrict__ vtr, _Float16* __restrict__ vti)
{
    const int lane = threadIdx.x & 31;
    const int wave = threadIdx.x >> 5;
    const int tile = blockIdx.x * 8 + wave;   // 16-row tile, 0..16383
    const int proj = blockIdx.y;
    const size_t m0 = (size_t)tile * 16;

    const float *Xr, *Xi, *Wr, *Wi;
    if (proj == 0)      { Xr = Qr;  Xi = Qi;  Wr = Wqr; Wi = Wqi; }
    else if (proj == 1) { Xr = KVr; Xi = KVi; Wr = Wkr; Wi = Wki; }
    else                { Xr = KVr; Xi = KVi; Wr = Wvr; Wi = Wvi; }

    v16h ar = load_a_f32(Xr + m0 * 32, 32, lane);
    v16h ai = load_a_f32(Xi + m0 * 32, 32, lane);
    v16h ain;   // -A_i (f16 WMMA NEG bits only cover C, so negate in-register)
#pragma unroll
    for (int i = 0; i < 16; ++i) ain[i] = -ai[i];

    const int mhi = (lane & 16) ? 8 : 0;
#pragma unroll
    for (int nt = 0; nt < 2; ++nt) {
        const int n0 = nt * 16;
        v16h br = load_b_w32(Wr, n0, lane);
        v16h bi = load_b_w32(Wi, n0, lane);
        v8f zero = {};
        v8f yr = wmma_f16(ar, br, wmma_f16(ain, bi, zero)); // Xr*Wr - Xi*Wi
        v8f yi = wmma_f16(ar, bi, wmma_f16(ai,  br, zero)); // Xr*Wi + Xi*Wr

        const int ncol = n0 + (lane & 15);
        if (proj < 2) {
            _Float16* outr = (proj == 0) ? qfr : kfr;
            _Float16* outi = (proj == 0) ? qfi : kfi;
#pragma unroll
            for (int r = 0; r < 8; ++r) {
                size_t m = m0 + r + mhi;
                outr[m * 32 + ncol] = (_Float16)yr[r];
                outi[m * 32 + ncol] = (_Float16)yi[r];
            }
        } else {
#pragma unroll
            for (int r = 0; r < 8; ++r) {
                size_t m = m0 + r + mhi;
                int b = (int)(m >> 14);          // / (S*E)
                int s = (int)(m >> 5) & (NS - 1);
                int x = (int)m & 31;
                int d = x * 32 + ncol;
                size_t idx = ((size_t)b * ND + d) * NS + s;  // [B,D,S]
                vtr[idx] = (_Float16)yr[r];
                vti[idx] = (_Float16)yi[r];
            }
        }
    }
}

// ---------------------------------------------------------------------------
// Kernel 2: scores[b,q,k] = qr.kr + qi.ki over D=1024.
// Wave computes a 16(q) x 64(k) strip: 32 d-steps x 8 WMMAs.
// ---------------------------------------------------------------------------
__global__ void score_kernel(const _Float16* __restrict__ qfr, const _Float16* __restrict__ qfi,
                             const _Float16* __restrict__ kfr, const _Float16* __restrict__ kfi,
                             float* __restrict__ scores)
{
    const int lane = threadIdx.x & 31;
    const int wave = threadIdx.x >> 5;
    const int id = blockIdx.x * 8 + wave;     // 0..4095
    const int b  = id >> 8;                   // 32 qtiles * 8 kgroups
    const int r  = id & 255;
    const int q0 = (r >> 3) * 16;
    const int k0c = (r & 7) * 64;

    const _Float16* qr_b = qfr + (size_t)b * NS * ND + (size_t)q0 * ND;
    const _Float16* qi_b = qfi + (size_t)b * NS * ND + (size_t)q0 * ND;
    const _Float16* kr_b = kfr + (size_t)b * NS * ND;
    const _Float16* ki_b = kfi + (size_t)b * NS * ND;

    v8f acc[4] = {};
    for (int d0 = 0; d0 < ND; d0 += 32) {
        v16h aR = load_a_f16(qr_b + d0, ND, lane);
        v16h aI = load_a_f16(qi_b + d0, ND, lane);
#pragma unroll
        for (int j = 0; j < 4; ++j) {
            const size_t koff = (size_t)(k0c + j * 16) * ND + d0;
            v16h bR = load_b_f16(kr_b + koff, ND, lane);
            v16h bI = load_b_f16(ki_b + koff, ND, lane);
            acc[j] = wmma_f16(aR, bR, acc[j]);
            acc[j] = wmma_f16(aI, bI, acc[j]);
        }
    }

    const int mhi = (lane & 16) ? 8 : 0;
    const int n   = lane & 15;
    float* sb = scores + (size_t)b * NS * NS;
#pragma unroll
    for (int j = 0; j < 4; ++j)
#pragma unroll
        for (int rr = 0; rr < 8; ++rr)
            sb[(size_t)(q0 + rr + mhi) * NS + (k0c + j * 16 + n)] = acc[j][rr];
}

// ---------------------------------------------------------------------------
// Kernel 3: masked softmax over k; one wave per (b,q) row; emits f16 probs.
// ---------------------------------------------------------------------------
__global__ void softmax_kernel(const float* __restrict__ scores,
                               const float* __restrict__ K_mask,
                               _Float16* __restrict__ probs)
{
    const int lane = threadIdx.x & 31;
    const int wave = threadIdx.x >> 5;
    const int row = blockIdx.x * 8 + wave;    // 0..B*S-1
    const int b = row >> 9;
    const float* src = scores + (size_t)row * NS;
    const float* km  = K_mask + (size_t)b * NS;

    float v[16];
    float mx = -3.0e38f;
#pragma unroll
    for (int i = 0; i < 16; ++i) {
        const int k = i * 32 + lane;
        float s = src[k] + (1.0f - km[k]) * -100000.0f;
        v[i] = s;
        mx = fmaxf(mx, s);
    }
#pragma unroll
    for (int off = 16; off > 0; off >>= 1) mx = fmaxf(mx, __shfl_xor(mx, off, 32));

    float sum = 0.0f;
#pragma unroll
    for (int i = 0; i < 16; ++i) { v[i] = __expf(v[i] - mx); sum += v[i]; }
#pragma unroll
    for (int off = 16; off > 0; off >>= 1) sum += __shfl_xor(sum, off, 32);
    const float inv = 1.0f / sum;

    _Float16* dst = probs + (size_t)row * NS;
#pragma unroll
    for (int i = 0; i < 16; ++i) dst[i * 32 + lane] = (_Float16)(v[i] * inv);
}

// ---------------------------------------------------------------------------
// Kernel 4: ctx = probs @ V (re & im), * Q_mask, interleaved complex output.
// Wave computes one 16(q) x 16(d) tile: 16 k-steps x 2 WMMAs.
// V is pre-transposed [B,D,S] so each lane's B-fragment is two 16B vector loads.
// ---------------------------------------------------------------------------
__global__ void ctx_kernel(const _Float16* __restrict__ probs,
                           const _Float16* __restrict__ vtr,
                           const _Float16* __restrict__ vti,
                           const float* __restrict__ Q_mask,
                           float* __restrict__ out)
{
    const int lane = threadIdx.x & 31;
    const int wave = threadIdx.x >> 5;
    const int id = blockIdx.x * 8 + wave;     // 0..32767
    const int b  = id >> 11;                  // 32 qtiles * 64 dtiles
    const int r  = id & 2047;
    const int q0 = (r >> 6) * 16;
    const int d0 = (r & 63) * 16;

    const _Float16* pb = probs + (size_t)b * NS * NS + (size_t)q0 * NS;
    const _Float16* vr = vtr + (size_t)b * ND * NS + (size_t)d0 * NS;
    const _Float16* vi = vti + (size_t)b * ND * NS + (size_t)d0 * NS;

    v8f accr = {}, acci = {};
    for (int k0 = 0; k0 < NS; k0 += 32) {
        v16h a  = load_a_f16(pb + k0, NS, lane);
        v16h br = load_b_f16(vr + k0, NS, lane);
        v16h bi = load_b_f16(vi + k0, NS, lane);
        accr = wmma_f16(a, br, accr);
        acci = wmma_f16(a, bi, acci);
    }

    const int mhi = (lane & 16) ? 8 : 0;
    const int n   = lane & 15;
    float2* o2 = reinterpret_cast<float2*>(out);
#pragma unroll
    for (int rr = 0; rr < 8; ++rr) {
        const int q = q0 + rr + mhi;
        const float qm = Q_mask[(size_t)b * NS + q];
        float2 val;
        val.x = accr[rr] * qm;
        val.y = acci[rr] * qm;
        o2[(size_t)(b * NS + q) * ND + (d0 + n)] = val;
    }
}

// ---------------------------------------------------------------------------
extern "C" void kernel_launch(void* const* d_in, const int* in_sizes, int n_in,
                              void* d_out, int out_size, void* d_ws, size_t ws_size,
                              hipStream_t stream)
{
    const float* Qr    = (const float*)d_in[0];
    const float* Qi    = (const float*)d_in[1];
    const float* KVr   = (const float*)d_in[2];
    const float* KVi   = (const float*)d_in[3];
    const float* Kmask = (const float*)d_in[4];
    const float* Qmask = (const float*)d_in[5];
    const float* Wqr   = (const float*)d_in[6];
    const float* Wqi   = (const float*)d_in[7];
    const float* Wkr   = (const float*)d_in[8];
    const float* Wki   = (const float*)d_in[9];
    const float* Wvr   = (const float*)d_in[10];
    const float* Wvi   = (const float*)d_in[11];

    const size_t nBSD = (size_t)NB * NS * ND;   // 8,388,608
    const size_t nBSS = (size_t)NB * NS * NS;   // 4,194,304

    char* ws = (char*)d_ws;
    _Float16* qfr = (_Float16*)ws; ws += nBSD * 2;
    _Float16* qfi = (_Float16*)ws; ws += nBSD * 2;
    _Float16* kfr = (_Float16*)ws; ws += nBSD * 2;
    _Float16* kfi = (_Float16*)ws; ws += nBSD * 2;
    _Float16* vtr = (_Float16*)ws; ws += nBSD * 2;   // V transposed [B,D,S]
    _Float16* vti = (_Float16*)ws; ws += nBSD * 2;
    float*    sco = (float*)ws;    ws += nBSS * 4;
    _Float16* prb = (_Float16*)ws; ws += nBSS * 2;   // total ~120 MB

    const dim3 blk(256);  // 8 wave32

    proj_kernel<<<dim3(2048, 3), blk, 0, stream>>>(Qr, Qi, KVr, KVi,
                                                   Wqr, Wqi, Wkr, Wki, Wvr, Wvi,
                                                   qfr, qfi, kfr, kfi, vtr, vti);
    score_kernel<<<dim3(512), blk, 0, stream>>>(qfr, qfi, kfr, kfi, sco);
    softmax_kernel<<<dim3(1024), blk, 0, stream>>>(sco, Kmask, prb);
    ctx_kernel<<<dim3(4096), blk, 0, stream>>>(prb, vtr, vti, Qmask, (float*)d_out);
}
